// TemporalCfCEncoder_80711025426843
// MI455X (gfx1250) — compile-verified, matrix operand order checked by reading, not run on hardware
//
#include <hip/hip_runtime.h>

typedef __bf16 bf16;
typedef __attribute__((ext_vector_type(16))) __bf16 v16bf;
typedef __attribute__((ext_vector_type(8)))  __bf16 v8bf;
typedef __attribute__((ext_vector_type(8)))  float   v8f;
typedef __attribute__((ext_vector_type(4)))  float   v4f;
typedef int v4i __attribute__((vector_size(16)));

#define TT 256
#define NN 256
#define DD 256
#define HH 4
#define DHE 64
#define LLY 2

// ---------------------------------------------------------------------------
// CDNA5 async global->LDS copy (ASYNCcnt-tracked), with portable fallback.
// Probe result: builtin takes (global v4i*, local v4i*, int offset, int cpol).
// ---------------------------------------------------------------------------
#if __has_builtin(__builtin_amdgcn_global_load_async_to_lds_b128) && \
    __has_builtin(__builtin_amdgcn_s_wait_asynccnt)
#define HAVE_ASYNC_LDS 1
#endif

__device__ __forceinline__ void copy16_g2l(const bf16* g, bf16* l) {
#ifdef HAVE_ASYNC_LDS
  __builtin_amdgcn_global_load_async_to_lds_b128(
      (__attribute__((address_space(1))) v4i*)g,
      (__attribute__((address_space(3))) v4i*)l, 0, 0);
#else
  *(v8bf*)l = *(const v8bf*)g;
#endif
}
__device__ __forceinline__ void async_drain() {
#ifdef HAVE_ASYNC_LDS
  __builtin_amdgcn_s_wait_asynccnt(0);
#endif
}

// ---------------------------------------------------------------------------
// WMMA fragment helpers (CDNA5 wave32 layouts, CDNA5 ISA §7.12.2)
// A 16x32 bf16 frag: lane = {m = lane&15 row, half = lane>>4};
//   element i -> K = i + half*8 + (i>=8 ? 8 : 0)  (two contiguous 8-elem runs)
// B 32x16 frag symmetric: lane&15 = output column = source row (K-contiguous).
// C 16x16 f32 frag: element r -> row = r + (lane>>4)*8, col = lane&15.
// ---------------------------------------------------------------------------
__device__ __forceinline__ v16bf load_frag_rm(const bf16* base, int lda, int r0, int k0) {
  int lane = threadIdx.x & 31;
  int m = lane & 15, half = lane >> 4;
  const bf16* p = base + (size_t)(r0 + m) * lda + k0 + half * 8;
  v8bf lo = *(const v8bf*)p;
  v8bf hi = *(const v8bf*)(p + 16);
  v16bf r;
#pragma unroll
  for (int i = 0; i < 8; ++i) { r[i] = lo[i]; r[i + 8] = hi[i]; }
  return r;
}

__device__ __forceinline__ v8f wmma_bf16(v16bf a, v16bf b, v8f c) {
  return __builtin_amdgcn_wmma_f32_16x16x32_bf16(false, a, false, b, (short)0, c, false, false);
}

// ---------------------------------------------------------------------------
// prep: dts[t] and re_bias[d]
// ---------------------------------------------------------------------------
__global__ __launch_bounds__(256) void prep_kernel(
    const float* __restrict__ st, const float* __restrict__ re_w,
    const float* __restrict__ re_b, const float* __restrict__ re_norm,
    float* __restrict__ dts, float* __restrict__ re_bias) {
  int i = threadIdx.x;
  dts[i] = (i == 0) ? st[0] : st[i] - st[i - 1];
  re_bias[i] = re_w[i] * re_norm[0] + re_b[i];
}

// ---------------------------------------------------------------------------
// elementwise f32 -> bf16 (weights once per launch; activations when needed)
// ---------------------------------------------------------------------------
__global__ __launch_bounds__(256) void f32_to_bf16_kernel(
    const float* __restrict__ s, bf16* __restrict__ d, int n4) {
  int i = blockIdx.x * 256 + threadIdx.x;
  int stride = gridDim.x * 256;
  for (int j = i; j < n4; j += stride) {
    v4f v = *(const v4f*)(s + (size_t)j * 4);
    d[(size_t)j * 4 + 0] = (bf16)v[0];
    d[(size_t)j * 4 + 1] = (bf16)v[1];
    d[(size_t)j * 4 + 2] = (bf16)v[2];
    d[(size_t)j * 4 + 3] = (bf16)v[3];
  }
}

// ---------------------------------------------------------------------------
// LayerNorm over D=256, one block per row, bf16 output
// ---------------------------------------------------------------------------
__global__ __launch_bounds__(256) void ln_kernel(
    const float* __restrict__ x, const float* __restrict__ g,
    const float* __restrict__ b, bf16* __restrict__ y) {
  __shared__ float red[256];
  size_t row = blockIdx.x;
  int d = threadIdx.x;
  float v = x[row * DD + d];
  red[d] = v;
  __syncthreads();
#pragma unroll
  for (int s = 128; s > 0; s >>= 1) {
    if (d < s) red[d] += red[d + s];
    __syncthreads();
  }
  float mean = red[0] * (1.0f / DD);
  __syncthreads();
  float c = v - mean;
  red[d] = c * c;
  __syncthreads();
#pragma unroll
  for (int s = 128; s > 0; s >>= 1) {
    if (d < s) red[d] += red[d + s];
    __syncthreads();
  }
  float var = red[0] * (1.0f / DD);
  float o = c * rsqrtf(var + 1e-5f) * g[d] + b[d];
  y[row * DD + d] = (bf16)o;
}

// ---------------------------------------------------------------------------
// Generic GEMM: C[M,Nc] = A(bf16)[M,K] x W(bf16)[Nc,K]^T + bias, epilogues:
//   EPI 0: bf16 store      EPI 1: f32 residual +=, optional extra vector
//   EPI 2: SiLU then bf16 store
// Block tile 32(M) x 128(N); 8 waves = 2(M) x 4(N); wave tile 16x32.
// A tile staged to LDS via async global->LDS copies.
// ---------------------------------------------------------------------------
template <int KDIM, int EPI>
__global__ __launch_bounds__(256) void gemm_kernel(
    const bf16* __restrict__ A, const bf16* __restrict__ W,
    const float* __restrict__ bias, const float* __restrict__ extra,
    bf16* __restrict__ outb, float* __restrict__ outf, int Nc) {
  __shared__ __align__(16) bf16 As[32 * (KDIM + 8)];
  const int tid = threadIdx.x;
  const size_t rowblk = (size_t)blockIdx.x * 32;
  const int colblk = blockIdx.y * 128;

  for (int i = tid * 8; i < 32 * KDIM; i += 256 * 8) {
    int r = i / KDIM, c = i % KDIM;
    copy16_g2l(&A[(rowblk + r) * (size_t)KDIM + c], &As[r * (KDIM + 8) + c]);
  }
  async_drain();
  __syncthreads();

  const int wave = tid >> 5, lane = tid & 31;
  const int wm = wave & 1, wn = wave >> 1;
  const int r0 = wm * 16;
  const int c0 = colblk + wn * 32;
  const int half = lane >> 4, nl = lane & 15;

  v8f acc[2] = {};
  for (int kb = 0; kb < KDIM; kb += 32) {
    v16bf a = load_frag_rm(As, KDIM + 8, r0, kb);
#pragma unroll
    for (int s = 0; s < 2; ++s) {
      v16bf bf = load_frag_rm(W, KDIM, c0 + s * 16, kb);
      acc[s] = wmma_bf16(a, bf, acc[s]);
    }
  }

#pragma unroll
  for (int s = 0; s < 2; ++s) {
#pragma unroll
    for (int e = 0; e < 8; ++e) {
      size_t row = rowblk + r0 + e + half * 8;
      int col = c0 + s * 16 + nl;
      float v = acc[s][e] + bias[col];
      if (EPI == 0) {
        outb[row * (size_t)Nc + col] = (bf16)v;
      } else if (EPI == 1) {
        float add = (extra != nullptr) ? extra[col] : 0.0f;
        outf[row * (size_t)Nc + col] += v + add;
      } else {
        v = v / (1.0f + __expf(-v));
        outb[row * (size_t)Nc + col] = (bf16)v;
      }
    }
  }
}

// ---------------------------------------------------------------------------
// Flash-style attention core: one block per (t, head), 8 waves x 32 rows.
// K rows loaded straight from global qkv (rows contiguous in d); V transposed
// into LDS once; streaming softmax with per-wave LDS scratch for P tiles.
// ---------------------------------------------------------------------------
__global__ __launch_bounds__(256) void attn_kernel(const bf16* __restrict__ qkv,
                                                   bf16* __restrict__ obf) {
  const int t = blockIdx.x >> 2;
  const int h = blockIdx.x & 3;
  __shared__ __align__(16) bf16 Vt[DHE * (NN + 8)];   // V^T: [64][264]
  __shared__ __align__(16) bf16 Ps[8 * 2 * 16 * 32];  // per-wave P scratch
  const int tid = threadIdx.x;
  const size_t qkvbase = (size_t)t * NN * 768;

  {  // transpose V into LDS: thread per source row m
    const bf16* vp = qkv + qkvbase + (size_t)tid * 768 + 512 + h * DHE;
#pragma unroll
    for (int d = 0; d < DHE; ++d) Vt[d * (NN + 8) + tid] = vp[d];
  }
  __syncthreads();

  const int wave = tid >> 5, lane = tid & 31;
  const int half = lane >> 4, nl = lane & 15;
  const int rbase = wave * 32;

  const bf16* qbase = qkv + qkvbase + h * DHE;
  const bf16* kbase = qkv + qkvbase + 256 + h * DHE;

  v16bf aq[2][2];
#pragma unroll
  for (int rt = 0; rt < 2; ++rt)
#pragma unroll
    for (int c = 0; c < 2; ++c)
      aq[rt][c] = load_frag_rm(qbase, 768, rbase + rt * 16, c * 32);

  v8f oacc[2][4] = {};
  float rowM[2][8], rowS[2][8];
#pragma unroll
  for (int rt = 0; rt < 2; ++rt)
#pragma unroll
    for (int e = 0; e < 8; ++e) { rowM[rt][e] = -1e30f; rowS[rt][e] = 0.0f; }

  bf16* myPs = Ps + wave * (2 * 16 * 32);

  for (int jb = 0; jb < NN; jb += 32) {
    v8f sa[2][2] = {};
#pragma unroll
    for (int c = 0; c < 2; ++c) {
      v16bf bk0 = load_frag_rm(kbase, 768, jb, c * 32);
      v16bf bk1 = load_frag_rm(kbase, 768, jb + 16, c * 32);
#pragma unroll
      for (int rt = 0; rt < 2; ++rt) {
        sa[rt][0] = wmma_bf16(aq[rt][c], bk0, sa[rt][0]);
        sa[rt][1] = wmma_bf16(aq[rt][c], bk1, sa[rt][1]);
      }
    }
#pragma unroll
    for (int rt = 0; rt < 2; ++rt) {
#pragma unroll
      for (int e = 0; e < 8; ++e) {
        float s0 = sa[rt][0][e] * 0.125f;
        float s1 = sa[rt][1][e] * 0.125f;
        float mx = fmaxf(s0, s1);
#pragma unroll
        for (int msk = 8; msk > 0; msk >>= 1) mx = fmaxf(mx, __shfl_xor(mx, msk, 32));
        float mnew = fmaxf(rowM[rt][e], mx);
        float scale = __expf(rowM[rt][e] - mnew);
        float p0 = __expf(s0 - mnew);
        float p1 = __expf(s1 - mnew);
        float sum = p0 + p1;
#pragma unroll
        for (int msk = 8; msk > 0; msk >>= 1) sum += __shfl_xor(sum, msk, 32);
        rowS[rt][e] = rowS[rt][e] * scale + sum;
        rowM[rt][e] = mnew;
#pragma unroll
        for (int vc = 0; vc < 4; ++vc) oacc[rt][vc][e] *= scale;
        int prow = e + half * 8;
        myPs[rt * 512 + prow * 32 + nl] = (bf16)p0;
        myPs[rt * 512 + prow * 32 + 16 + nl] = (bf16)p1;
      }
    }
    // O += P @ V  (per-wave LDS round-trip reshapes C-frag -> A-frag)
#pragma unroll
    for (int rt = 0; rt < 2; ++rt) {
      v16bf ap = load_frag_rm(myPs + rt * 512, 32, 0, 0);
#pragma unroll
      for (int vc = 0; vc < 4; ++vc) {
        v16bf bv = load_frag_rm(Vt, NN + 8, vc * 16, jb);
        oacc[rt][vc] = wmma_bf16(ap, bv, oacc[rt][vc]);
      }
    }
  }

#pragma unroll
  for (int rt = 0; rt < 2; ++rt)
#pragma unroll
    for (int vc = 0; vc < 4; ++vc)
#pragma unroll
      for (int e = 0; e < 8; ++e) {
        int row = rbase + rt * 16 + e + half * 8;
        int col = vc * 16 + nl;
        float v = oacc[rt][vc][e] / rowS[rt][e];
        obf[((size_t)t * NN + row) * DD + h * DHE + col] = (bf16)v;
      }
}

// ---------------------------------------------------------------------------
// One CfC timestep, bf16 recurrence: xh = [x_t | h_{t-1}] staged to LDS via
// async copies (both halves already bf16). Three fused GEMMs share the A tile.
// Writes bf16 h (next step / next layer input); optional f32 copy to d_out.
// Grid (8,2): block tile 32 rows x 128 cols; 8 waves = 2(M) x 4(N).
// ---------------------------------------------------------------------------
__global__ __launch_bounds__(256) void cfc_step_kernel(
    const bf16* __restrict__ xseq, const bf16* __restrict__ hprev,
    bf16* __restrict__ hout, float* __restrict__ hout_f32,
    const bf16* __restrict__ W1, const float* __restrict__ b1,
    const bf16* __restrict__ W2, const float* __restrict__ b2,
    const bf16* __restrict__ Wt, const float* __restrict__ bt,
    const float* __restrict__ log_tau, const float* __restrict__ dts, int t) {
  __shared__ __align__(16) bf16 Xs[32 * 520];
  const int tid = threadIdx.x;
  const int rowblk = blockIdx.x * 32;
  const int colblk = blockIdx.y * 128;

  // 2048 16B chunks: 64 per row (512 bf16 = x[256] | h[256])
  for (int ch = tid; ch < 2048; ch += 256) {
    int r = ch >> 6;
    int c = (ch & 63) * 8;
    bf16* dst = &Xs[r * 520 + c];
    if (c < 256) {
      copy16_g2l(&xseq[(size_t)(rowblk + r) * DD + c], dst);
    } else if (hprev != nullptr) {
      copy16_g2l(&hprev[(size_t)(rowblk + r) * DD + (c - 256)], dst);
    } else {
      v8bf z = {};
      *(v8bf*)dst = z;
    }
  }
  async_drain();
  __syncthreads();

  const int wave = tid >> 5, lane = tid & 31;
  const int wm = wave & 1, wn = wave >> 1;
  const int r0 = wm * 16;
  const int c0 = colblk + wn * 32;
  const int half = lane >> 4, nl = lane & 15;

  v8f f1a[2] = {}, f2a[2] = {}, tba[2] = {};
  for (int kb = 0; kb < 512; kb += 32) {
    v16bf a = load_frag_rm(Xs, 520, r0, kb);
#pragma unroll
    for (int s = 0; s < 2; ++s) {
      int n0 = c0 + s * 16;
      f1a[s] = wmma_bf16(a, load_frag_rm(W1, 512, n0, kb), f1a[s]);
      f2a[s] = wmma_bf16(a, load_frag_rm(W2, 512, n0, kb), f2a[s]);
      tba[s] = wmma_bf16(a, load_frag_rm(Wt, 512, n0, kb), tba[s]);
    }
  }

  float dt = dts[t];
#pragma unroll
  for (int s = 0; s < 2; ++s) {
#pragma unroll
    for (int e = 0; e < 8; ++e) {
      int row = rowblk + r0 + e + half * 8;
      int col = c0 + s * 16 + nl;
      float tau = __expf(log_tau[col]);
      float f1 = tanhf(f1a[s][e] + b1[col]);
      float f2 = tanhf(f2a[s][e] + b2[col]);
      float tb = tba[s][e] + bt[col];
      float g = 1.0f / (1.0f + __expf(tau * dt - tb));
      float hv = g * f1 + (1.0f - g) * f2;
      hout[(size_t)row * DD + col] = (bf16)hv;
      if (hout_f32 != nullptr) hout_f32[(size_t)row * DD + col] = hv;
    }
  }
}

// ---------------------------------------------------------------------------
extern "C" void kernel_launch(void* const* d_in, const int* in_sizes, int n_in,
                              void* d_out, int out_size, void* d_ws, size_t ws_size,
                              hipStream_t stream) {
  (void)in_sizes; (void)n_in; (void)out_size; (void)ws_size;
  const float* x_in    = (const float*)d_in[0];
  const float* re_norm = (const float*)d_in[1];
  const float* stime   = (const float*)d_in[2];
  const float* re_w    = (const float*)d_in[3];
  const float* re_b    = (const float*)d_in[4];
  const float* ln1_g   = (const float*)d_in[5];
  const float* ln1_b   = (const float*)d_in[6];
  const float* in_w    = (const float*)d_in[7];
  const float* in_b    = (const float*)d_in[8];
  const float* out_w   = (const float*)d_in[9];
  const float* out_b   = (const float*)d_in[10];
  const float* ln2_g   = (const float*)d_in[11];
  const float* ln2_b   = (const float*)d_in[12];
  const float* ffa_w   = (const float*)d_in[13];
  const float* ffa_b   = (const float*)d_in[14];
  const float* ffb_w   = (const float*)d_in[15];
  const float* ffb_b   = (const float*)d_in[16];
  const float* c1w     = (const float*)d_in[17];
  const float* c1b     = (const float*)d_in[18];
  const float* c2w     = (const float*)d_in[19];
  const float* c2b     = (const float*)d_in[20];
  const float* lta     = (const float*)d_in[21];
  const float* tbw     = (const float*)d_in[22];
  const float* tbb     = (const float*)d_in[23];

  char* ws = (char*)d_ws;
  constexpr size_t SZ = (size_t)TT * NN * DD * 2;   // one [T,N,D] bf16 tensor
  constexpr size_t SZ4 = (size_t)TT * NN * DD * 4;  // one [T,N,D] f32 tensor
  float* dts    = (float*)(ws);
  float* rebias = (float*)(ws + 1024);
  bf16*  ybf    = (bf16*)(ws + 4096);            // LN out; later layer-1 x (bf16)
  bf16*  qkv    = (bf16*)(ws + 4096 + SZ);
  bf16*  obf    = (bf16*)(ws + 4096 + SZ * 4);   // attn out; later layer-1 h-seq
  bf16*  hbf    = (bf16*)(ws + 4096 + SZ * 5);   // FFN hidden; later layer-2 h-seq
  float* xr     = (float*)(ws + 4096 + SZ * 7);
  char*  wpool  = ws + 4096 + SZ * 7 + SZ4;      // bf16 weight pool (~2.5MB)
  bf16* in_wb  = (bf16*)(wpool);                                  // 768*256
  bf16* out_wb = in_wb + 768 * 256;                               // 256*256
  bf16* ffa_wb = out_wb + 256 * 256;                              // 512*256
  bf16* ffb_wb = ffa_wb + 512 * 256;                              // 256*512
  bf16* c1wb   = ffb_wb + 256 * 512;                              // 2*256*512
  bf16* c2wb   = c1wb + 2 * 256 * 512;
  bf16* tbwb   = c2wb + 2 * 256 * 512;

  // one-shot weight conversion to bf16 (stays resident in L2 afterwards)
  auto cvt = [&](const float* s, bf16* d, int n) {
    int n4 = n / 4;
    int blocks = (n4 + 255) / 256;
    f32_to_bf16_kernel<<<blocks, 256, 0, stream>>>(s, d, n4);
  };
  cvt(in_w, in_wb, 768 * 256);
  cvt(out_w, out_wb, 256 * 256);
  cvt(ffa_w, ffa_wb, 512 * 256);
  cvt(ffb_w, ffb_wb, 256 * 512);
  cvt(c1w, c1wb, 2 * 256 * 512);
  cvt(c2w, c2wb, 2 * 256 * 512);
  cvt(tbw, tbwb, 2 * 256 * 512);

  (void)hipMemcpyAsync(xr, x_in, SZ4, hipMemcpyDeviceToDevice, stream);
  prep_kernel<<<1, 256, 0, stream>>>(stime, re_w, re_b, re_norm, dts, rebias);

  // token self-attention block
  ln_kernel<<<TT * NN, 256, 0, stream>>>(xr, ln1_g, ln1_b, ybf);
  gemm_kernel<256, 0><<<dim3(2048, 6), 256, 0, stream>>>(ybf, in_wb, in_b, nullptr, qkv, nullptr, 768);
  attn_kernel<<<TT * HH, 256, 0, stream>>>(qkv, obf);
  gemm_kernel<256, 1><<<dim3(2048, 2), 256, 0, stream>>>(obf, out_wb, out_b, nullptr, nullptr, xr, 256);
  ln_kernel<<<TT * NN, 256, 0, stream>>>(xr, ln2_g, ln2_b, ybf);
  gemm_kernel<256, 2><<<dim3(2048, 4), 256, 0, stream>>>(ybf, ffa_wb, ffa_b, nullptr, hbf, nullptr, 512);
  // last FFN projection folds residual + re_bias
  gemm_kernel<512, 1><<<dim3(2048, 2), 256, 0, stream>>>(hbf, ffb_wb, ffb_b, rebias, nullptr, xr, 256);

  // token-block output -> bf16 sequence for CfC layer 1
  f32_to_bf16_kernel<<<16384, 256, 0, stream>>>(xr, ybf, (TT * NN * DD) / 4);

  // two sequential CfC layers, recurrence carried in bf16
  float* outp = (float*)d_out;
  for (int l = 0; l < LLY; ++l) {
    const bf16* xseq = (l == 0) ? ybf : obf;
    bf16* hseq = (l == 0) ? obf : hbf;
    const bf16* W1 = c1wb + (size_t)l * DD * 2 * DD;
    const bf16* W2 = c2wb + (size_t)l * DD * 2 * DD;
    const bf16* Wt = tbwb + (size_t)l * DD * 2 * DD;
    for (int t = 0; t < TT; ++t) {
      const bf16* hp = (t == 0) ? nullptr : hseq + (size_t)(t - 1) * NN * DD;
      float* hf = (l == LLY - 1) ? outp + (size_t)t * NN * DD : nullptr;
      cfc_step_kernel<<<dim3(8, 2), 256, 0, stream>>>(
          xseq + (size_t)t * NN * DD, hp, hseq + (size_t)t * NN * DD, hf,
          W1, c1b + l * DD, W2, c2b + l * DD, Wt, tbb + l * DD,
          lta + l * DD, dts, t);
    }
  }
}
